// LayerStacks_47974784696704
// MI455X (gfx1250) — matrix operand order repaired and checked by reading, not compiled, not run on hardware
//
#include <hip/hip_runtime.h>

typedef __attribute__((ext_vector_type(16))) _Float16 v16h;
typedef __attribute__((ext_vector_type(8)))  _Float16 v8h;
typedef __attribute__((ext_vector_type(8)))  float    v8f;

#define WAVES 8
#define TPW   8   // tiles (of 16 samples) per wave; WG covers 8*8*16 = 1024 samples

// All weights staged in LDS, WMMA B-operands pre-permuted to lane order.
struct SmemLayout {
  _Float16 pool1[24 * 512];   // L1 B operands: (pair q, mat t, kchunk j) x 32 lanes x 16 halves
  _Float16 pool2[24 * 512];   // L2 B operands: (bucket c, outgroup g)   x 32 lanes x 16 halves
  _Float16 wout [6 * 320];    // Wout as f16
  float wmob_b [48];          // W1b[c][o][128]  (mobility column)
  float b1b_t  [48];
  float wmob_pa[48];
  float b1pa_t [48];
  float b2t    [384];
  float bout_t [6];
  float lpart  [WAVES * 32];            // per-lane Lout partials (x contribution)
  _Float16 l1buf[WAVES * 16 * 32];      // selected l1c activations, [m][k]
  _Float16 l2buf[WAVES * 16 * 64];      // selected l2 activations,  [m][o]
};

__device__ __forceinline__ float clamp01f(float v) { return fminf(fmaxf(v, 0.0f), 1.0f); }

__device__ __forceinline__ v16h load16h_lds(const _Float16* p) {
  v8h a = *(const v8h*)(p);
  v8h b = *(const v8h*)(p + 8);
  v16h r;
#pragma unroll
  for (int h = 0; h < 8; ++h) { r[h] = a[h]; r[h + 8] = b[h]; }
  return r;
}

__global__ __launch_bounds__(256) void layerstacks_kernel(
    const float* __restrict__ x_base, const float* __restrict__ x_pa,
    const float* __restrict__ mobility, const int* __restrict__ ply,
    const float* __restrict__ W1b,  const float* __restrict__ b1b,
    const float* __restrict__ W1pa, const float* __restrict__ b1pa,
    const float* __restrict__ W2,   const float* __restrict__ b2,
    const float* __restrict__ Wout, const float* __restrict__ bout,
    float* __restrict__ out, int B)
{
  extern __shared__ char smem_raw[];
  SmemLayout* S = reinterpret_cast<SmemLayout*>(smem_raw);
  const int tid = threadIdx.x;

  // ---------------- one-time cooperative weight staging ----------------
  // L1 B pool. Operand op = (q*2+t)*4 + j. Lane L, half h:
  //   n = L&15 (column), hi = L>>4; bucket c = 2q + (n>>3); neuron o = n&7;
  //   K(local) = (h < 8 ? 8*hi + h : 16 + 8*hi + h - 8), global k = 32*j + K(local)   [ISA 16-bit B K x 16 layout]
  for (int i = tid; i < 24 * 512; i += 256) {
    int h = i & 15, L = (i >> 4) & 31, op = i >> 9;
    int j = op & 3, t = (op >> 2) & 1, q = op >> 3;
    int n = L & 15, hi = L >> 4;
    int c = 2 * q + (n >> 3), o = n & 7;
    int k = 32 * j + (h < 8 ? 8 * hi + h : 16 + 8 * hi + (h - 8));
    const float* W = t ? W1pa : W1b;
    S->pool1[i] = (_Float16)W[(c * 8 + o) * 129 + k];
  }
  // L2 B pool. Operand r = c*4 + g; column n -> neuron o = 16g+n; k = (h < 8 ? 8*hi + h : 16 + 8*hi + h - 8) (K=32).
  for (int i = tid; i < 24 * 512; i += 256) {
    int h = i & 15, L = (i >> 4) & 31, r = i >> 9;
    int c = r >> 2, g = r & 3;
    int n = L & 15, hi = L >> 4;
    int o = 16 * g + n, k = (h < 8 ? 8 * hi + h : 16 + 8 * hi + (h - 8));
    S->pool2[i] = (_Float16)W2[(c * 64 + o) * 32 + k];
  }
  for (int i = tid; i < 6 * 320; i += 256) S->wout[i] = (_Float16)Wout[i];
  for (int i = tid; i < 48; i += 256) {
    int c = i >> 3, o = i & 7;
    S->wmob_b[i]  = W1b [(c * 8 + o) * 129 + 128];
    S->wmob_pa[i] = W1pa[(c * 8 + o) * 129 + 128];
    S->b1b_t[i]   = b1b[i];
    S->b1pa_t[i]  = b1pa[i];
  }
  for (int i = tid; i < 384; i += 256) S->b2t[i] = b2[i];
  if (tid < 6) S->bout_t[tid] = bout[tid];
  __syncthreads();

  const int wv = tid >> 5, lane = tid & 31;
  const int hi = lane >> 4, n = lane & 15;   // n: A row / B,D column for this lane
  _Float16* l1w = S->l1buf + wv * (16 * 32);
  _Float16* l2w = S->l2buf + wv * (16 * 64);
  float*    lp  = S->lpart + wv * 32;

  for (int tl = 0; tl < TPW; ++tl) {
    const int  T  = blockIdx.x * (WAVES * TPW) + wv * TPW + tl;
    const long s0 = (long)T * 16;
    if (s0 + 16 > (long)B) continue;          // wave-uniform guard

    // per-lane metadata
    const int cm = ply[s0 + n] / 10;          // bucket of the sample this lane owns as A-row / finalizer
    int   idxs[8];
    float mobv[8];
#pragma unroll
    for (int r = 0; r < 8; ++r) {             // buckets/mobility of D-rows m = 8*hi + r
      int m = 8 * hi + r;
      idxs[r] = ply[s0 + m] / 10;
      mobv[r] = fminf(mobility[s0 + m] * (7.0f / 255.0f), 1.0f);
    }

    // ---- load x into WMMA A layout (f16) + accumulate fp32 Lout x-partials ----
    float lpartial = 0.0f;
    v16h A[2][4];
#pragma unroll
    for (int t = 0; t < 2; ++t) {
      const float* X = (t ? x_pa : x_base) + (s0 + n) * 128;
      const _Float16* wrow = S->wout + cm * 320 + 64 + 128 * t;   // feats: [l2x(64), x_base(128), x_pa(128)]
#pragma unroll
      for (int j = 0; j < 4; ++j) {
        const int b0 = 32 * j + 8 * hi;
        float4 f0 = *(const float4*)(X + b0);
        float4 f1 = *(const float4*)(X + b0 + 4);
        float4 f2 = *(const float4*)(X + b0 + 16);
        float4 f3 = *(const float4*)(X + b0 + 20);
        float xs[16] = {f0.x, f0.y, f0.z, f0.w, f1.x, f1.y, f1.z, f1.w,
                        f2.x, f2.y, f2.z, f2.w, f3.x, f3.y, f3.z, f3.w};
#pragma unroll
        for (int h = 0; h < 16; ++h) {
          int k = b0 + (h < 8 ? h : h + 8);    // ISA 16-bit A 16x32 half->K map
          lpartial += xs[h] * (float)wrow[k];
          A[t][j][h] = (_Float16)xs[h];
        }
      }
    }

    // ---- L1: 3 bucket-pair passes x {base,pa} x 4 k-chunks = 24 WMMAs ----
#pragma unroll
    for (int t = 0; t < 2; ++t) {
      float selV[8];
#pragma unroll
      for (int r = 0; r < 8; ++r) selV[r] = 0.0f;
#pragma unroll
      for (int q = 0; q < 3; ++q) {
        const int op = (q * 2 + t) * 4;
        v8f acc = {};
#pragma unroll
        for (int j = 0; j < 4; ++j) {
          v16h Bm = load16h_lds(S->pool1 + (op + j) * 512 + lane * 16);
          acc = __builtin_amdgcn_wmma_f32_16x16x32_f16(false, A[t][j], false, Bm,
                                                       (short)0, acc, false, false);
        }
        const int c = 2 * q + (n >> 3), o = n & 7;
        const float wm = (t ? S->wmob_pa : S->wmob_b)[c * 8 + o];
        const float bb = (t ? S->b1pa_t  : S->b1b_t )[c * 8 + o];
#pragma unroll
        for (int r = 0; r < 8; ++r) {
          float v = acc[r] + wm * mobv[r] + bb;     // mobility column + bias fixup
          selV[r] = (idxs[r] == c) ? v : selV[r];
        }
      }
      const int o = n & 7;
#pragma unroll
      for (int r = 0; r < 8; ++r) {
        // column-parity ownership: lane's bucket parity (n>>3) must match sample's bucket parity
        if ((idxs[r] & 1) == (n >> 3)) {
          const int m = 8 * hi + r;
          float v   = selV[r];
          float sq  = fminf(v * v * (255.0f / 256.0f), 1.0f);
          float lin = clamp01f(v);
          l1w[m * 32 + 8 * t + o]      = (_Float16)sq;   // l1c K: [b_sq, pa_sq, b_lin, pa_lin]
          l1w[m * 32 + 16 + 8 * t + o] = (_Float16)lin;
        }
      }
    }
    __builtin_amdgcn_wave_barrier();

    // ---- build A2 (16 samples x 32 l1c) from LDS ----
    v16h A2;
    {
      const _Float16* lrow = l1w + n * 32;
      v8h lo = *(const v8h*)(lrow + 8 * hi);
      v8h hh = *(const v8h*)(lrow + 16 + 8 * hi);
#pragma unroll
      for (int h = 0; h < 8; ++h) { A2[h] = lo[h]; A2[h + 8] = hh[h]; }
    }

    // ---- L2: 6 buckets x 4 output groups = 24 WMMAs (K=32 in one step) ----
#pragma unroll
    for (int g = 0; g < 4; ++g) {
      float accSel[8];
#pragma unroll
      for (int r = 0; r < 8; ++r) accSel[r] = 0.0f;
#pragma unroll
      for (int c = 0; c < 6; ++c) {
        v16h Bm = load16h_lds(S->pool2 + (c * 4 + g) * 512 + lane * 16);
        v8f z = {};
        v8f acc = __builtin_amdgcn_wmma_f32_16x16x32_f16(false, A2, false, Bm,
                                                         (short)0, z, false, false);
        const float bb = S->b2t[c * 64 + 16 * g + n];
#pragma unroll
        for (int r = 0; r < 8; ++r) {
          float v   = acc[r] + bb;
          float a1  = clamp01f(v);
          float act = a1 * a1 * (255.0f / 256.0f);
          accSel[r] = (idxs[r] == c) ? act : accSel[r];
        }
      }
#pragma unroll
      for (int r = 0; r < 8; ++r) {
        const int m = 8 * hi + r;
        l2w[m * 64 + 16 * g + n] = (_Float16)accSel[r];
      }
    }

    lp[lane] = lpartial;
    __builtin_amdgcn_wave_barrier();

    // ---- Lout finalize: lanes 0..15 own one sample each ----
    if (hi == 0) {
      const int m = n, c = cm;
      float sum = lp[m] + lp[m + 16] + S->bout_t[c];
      const _Float16* l2r = l2w + m * 64;
      const _Float16* wr  = S->wout + c * 320;   // l2x features occupy Wout[0..63]
#pragma unroll 8
      for (int o = 0; o < 64; ++o) sum += (float)l2r[o] * (float)wr[o];
      out[s0 + m] = sum;
    }
  }
}

extern "C" void kernel_launch(void* const* d_in, const int* in_sizes, int n_in,
                              void* d_out, int out_size, void* d_ws, size_t ws_size,
                              hipStream_t stream) {
  const float* x_base   = (const float*)d_in[0];
  const float* x_pa     = (const float*)d_in[1];
  const float* mobility = (const float*)d_in[2];
  const int*   ply      = (const int*)d_in[3];
  const float* W1b      = (const float*)d_in[4];
  const float* b1b      = (const float*)d_in[5];
  const float* W1pa     = (const float*)d_in[6];
  const float* b1pa     = (const float*)d_in[7];
  const float* W2       = (const float*)d_in[8];
  const float* b2       = (const float*)d_in[9];
  const float* Wout     = (const float*)d_in[10];
  const float* bout     = (const float*)d_in[11];
  float* out = (float*)d_out;

  const int B = in_sizes[3];                 // ply has B elements
  const int samplesPerWG = WAVES * TPW * 16; // 1024
  const int nWG = (B + samplesPerWG - 1) / samplesPerWG;
  const size_t smem = sizeof(SmemLayout);    // ~81 KB dynamic LDS (<= 320 KB/WG on gfx1250)

  layerstacks_kernel<<<nWG, 256, smem, stream>>>(
      x_base, x_pa, mobility, ply, W1b, b1b, W1pa, b1pa, W2, b2, Wout, bout, out, B);
}
